// PhysicsGraphNeuralODEFunc_39754217292306
// MI455X (gfx1250) — matrix operand order repaired and checked by reading, not compiled, not run on hardware
//
#include <hip/hip_runtime.h>
#include <hip/hip_bf16.h>

typedef __attribute__((ext_vector_type(16))) _Float16 v16h;
typedef __attribute__((ext_vector_type(8)))  _Float16 v8h;
typedef __attribute__((ext_vector_type(8)))  float    v8f;
typedef __attribute__((ext_vector_type(4)))  float    v4f;
typedef __attribute__((ext_vector_type(4)))  int      v4i;

#define B_ROWS 16384
#define DDIM   512
#define HIDN   512
#define NH     1024      // quad + cubic hidden concatenated
#define EHN    32
#define OMEGA_F 0.5235987755982988f   // 2*pi/12

// ---------------- CDNA5 async global->LDS copy path ----------------
#if __has_builtin(__builtin_amdgcn_global_load_async_to_lds_b128)
#define HAVE_ASYNC_LDS 1
#else
#define HAVE_ASYNC_LDS 0
#endif

__device__ __forceinline__ void cp16_to_lds(void* lds_dst, const void* gsrc) {
#if HAVE_ASYNC_LDS
  // GLOBAL_LOAD_ASYNC_TO_LDS_B128: per-lane 16B, tracked by ASYNCcnt
  __builtin_amdgcn_global_load_async_to_lds_b128(
      (__attribute__((address_space(1))) v4i*)(gsrc),
      (__attribute__((address_space(3))) v4i*)(lds_dst),
      0, 0);
#else
  *(v8h*)lds_dst = *(const v8h*)gsrc;
#endif
}

template <int N>
__device__ __forceinline__ void wait_asynccnt() {
#if HAVE_ASYNC_LDS
#if __has_builtin(__builtin_amdgcn_s_wait_asynccnt)
  __builtin_amdgcn_s_wait_asynccnt(N);
#else
  asm volatile("s_wait_asynccnt %0" :: "i"(N) : "memory");
#endif
#endif
}

// ---------------- prep kernels ----------------

// Lh[i*512+j] = L[i,j] in f16; row-major layout IS the [N][K] operand for
// out[b,i] = sum_j x[b,j] * L[i,j].
__global__ void k_build_L(const float* __restrict__ fc, const float* __restrict__ t,
                          _Float16* __restrict__ Lh) {
  int idx = blockIdx.x * blockDim.x + threadIdx.x;     // over 512*512
  if (idx >= DDIM * DDIM) return;
  float ts = t[0];
  float c1 = __cosf(OMEGA_F * ts),        s1 = __sinf(OMEGA_F * ts);
  float c2 = __cosf(2.0f * OMEGA_F * ts), s2 = __sinf(2.0f * OMEGA_F * ts);
  const float* p = fc + (size_t)idx * 5;
  float v = p[0] + p[1] * c1 + p[2] * s1 + p[3] * c2 + p[4] * s2;
  Lh[idx] = (_Float16)v;
}

// WT[h*512+d] = W1[d,h] (f16, [N][K] layout), bb = concat bias, vv = rowmean(W2)
__global__ void k_prep_weights(const float* __restrict__ qW1, const float* __restrict__ cW1,
                               const float* __restrict__ qb1, const float* __restrict__ cb1,
                               const float* __restrict__ qW2, const float* __restrict__ cW2,
                               _Float16* __restrict__ WT, float* __restrict__ bb,
                               float* __restrict__ vv) {
  int idx = blockIdx.x * blockDim.x + threadIdx.x;
  if (idx < NH * DDIM) {
    int h = idx / DDIM;
    int d = idx - h * DDIM;
    float w = (h < HIDN) ? qW1[(size_t)d * HIDN + h] : cW1[(size_t)d * HIDN + (h - HIDN)];
    WT[idx] = (_Float16)w;
  }
  if (idx < NH) {
    bb[idx] = (idx < HIDN) ? qb1[idx] : cb1[idx - HIDN];
    const float* r = (idx < HIDN) ? (qW2 + (size_t)idx * DDIM)
                                  : (cW2 + (size_t)(idx - HIDN) * DDIM);
    float sacc = 0.0f;
    for (int i = 0; i < DDIM; ++i) sacc += r[i];
    vv[idx] = sacc * (1.0f / DDIM);
  }
}

__global__ void k_mb2(const float* __restrict__ qb2, const float* __restrict__ cb2,
                      float* __restrict__ mb2) {
  __shared__ float red[256];
  float a = 0.0f;
  for (int i = threadIdx.x; i < DDIM; i += 256) a += qb2[i] + cb2[i];
  red[threadIdx.x] = a;
  __syncthreads();
  for (int s = 128; s > 0; s >>= 1) {
    if ((int)threadIdx.x < s) red[threadIdx.x] += red[threadIdx.x + s];
    __syncthreads();
  }
  if (threadIdx.x == 0) mb2[0] = red[0] * (1.0f / DDIM);
}

__global__ void k_cvt_x(const float* __restrict__ x, _Float16* __restrict__ xh) {
  int idx = blockIdx.x * blockDim.x + threadIdx.x;   // each thread converts 8 elems
  if (idx * 8 >= B_ROWS * DDIM) return;
  const v4f* xp = (const v4f*)x;
  v4f a = xp[(size_t)idx * 2 + 0];
  v4f b = xp[(size_t)idx * 2 + 1];
  v8h o;
  o[0] = (_Float16)a[0]; o[1] = (_Float16)a[1]; o[2] = (_Float16)a[2]; o[3] = (_Float16)a[3];
  o[4] = (_Float16)b[0]; o[5] = (_Float16)b[1]; o[6] = (_Float16)b[2]; o[7] = (_Float16)b[3];
  ((v8h*)xh)[idx] = o;
}

// ENSO heads: one wave per row, lane = hidden unit.
__global__ void k_enso(const float* __restrict__ x,
                       const float* __restrict__ tW1, const float* __restrict__ tb1,
                       const float* __restrict__ tW2, const float* __restrict__ tb2,
                       const float* __restrict__ hW1, const float* __restrict__ hb1,
                       const float* __restrict__ hW2, const float* __restrict__ hb2,
                       float* __restrict__ c01) {
  int lane = threadIdx.x & 31;
  int row  = blockIdx.x * 8 + (threadIdx.x >> 5);
  if (row >= B_ROWS) return;
  float T = x[(size_t)row * DDIM + 0];
  float H = x[(size_t)row * DDIM + 1];
  float f0 = T, f1 = H, f2 = T * T, f3 = T * H;
  float fT4 = T * T * T, fH4 = T * H * H;
  float hT = tb1[lane] + f0 * tW1[0 * EHN + lane] + f1 * tW1[1 * EHN + lane]
           + f2 * tW1[2 * EHN + lane] + f3 * tW1[3 * EHN + lane] + fT4 * tW1[4 * EHN + lane];
  float cT = fmaxf(hT, 0.0f) * tW2[lane];
  float hH = hb1[lane] + f0 * hW1[0 * EHN + lane] + f1 * hW1[1 * EHN + lane]
           + f2 * hW1[2 * EHN + lane] + f3 * hW1[3 * EHN + lane] + fH4 * hW1[4 * EHN + lane];
  float cH = fmaxf(hH, 0.0f) * hW2[lane];
  for (int o = 16; o >= 1; o >>= 1) {
    cT += __shfl_xor(cT, o, 32);
    cH += __shfl_xor(cH, o, 32);
  }
  if (lane == 0) {
    c01[(size_t)row * 2 + 0] = cT + tb2[0];
    c01[(size_t)row * 2 + 1] = cH + hb2[0];
  }
}

// ---------------- WMMA GEMM: C[M x N] = A[M x 512] * Bw[N x 512]^T ----------------
// Double-buffered LDS, async global->LDS tile loads overlapped with WMMA compute.
// FINAL==0: relu(acc+bias)*v, reduce over N, atomicAdd into sbuf[row]
// FINAL==1: acc + sbuf[row] + mb2 (+ ENSO cols 0/1) -> out (f32)
template <int FINAL>
__global__ __launch_bounds__(256)
void k_gemm(const _Float16* __restrict__ A,
            const _Float16* __restrict__ Bw,
            const float* __restrict__ bias,
            const float* __restrict__ vvec,
            float* __restrict__ sbuf,
            const float* __restrict__ mb2,
            const float* __restrict__ c01,
            float* __restrict__ out) {
  __shared__ __align__(16) _Float16 As[2][128][40];
  __shared__ __align__(16) _Float16 Bs[2][128][40];

  const int tid  = threadIdx.x;
  const int lane = tid & 31;
  const int wid  = tid >> 5;
  const int wm   = wid & 3;          // 32-row band within 128
  const int wn   = wid >> 2;         // 64-col band within 128
  const int m0   = blockIdx.y * 128;
  const int n0   = blockIdx.x * 128;
  const int hi   = lane >> 4;        // lane half
  const int l15  = lane & 15;

  v8f acc[2][4] = {};

  // Each thread moves 4 x 16B per k-step: 2 chunks of A, 2 chunks of B.
  auto issue_tiles = [&](int buf, int kk) {
    #pragma unroll
    for (int c = tid; c < 512; c += 256) {
      int row = c >> 2, seg = (c & 3) * 8;
      cp16_to_lds(&As[buf][row][seg], A  + (size_t)(m0 + row) * 512 + kk + seg);
      cp16_to_lds(&Bs[buf][row][seg], Bw + (size_t)(n0 + row) * 512 + kk + seg);
    }
  };

  issue_tiles(0, 0);

  for (int step = 0; step < 16; ++step) {
    const int cur = step & 1;
    __syncthreads();                       // everyone done reading buffer cur^1
    if (step + 1 < 16) {
      issue_tiles(cur ^ 1, (step + 1) * 32);
      wait_asynccnt<4>();                  // in-order: <=4 outstanding => cur done
    } else {
      wait_asynccnt<0>();
    }
    __syncthreads();                       // all waves' cur tiles visible

    v16h af[2], bf[4];
    // A 16x32 f16 layout: lane half 'hi' holds K chunks [hi*8..] and [16+hi*8..]
    #pragma unroll
    for (int i = 0; i < 2; ++i) {
      const _Float16* ar = &As[cur][wm * 32 + i * 16 + l15][0];
      v8h alo = *(const v8h*)(ar + hi * 8);
      v8h ahi = *(const v8h*)(ar + 16 + hi * 8);
      af[i] = __builtin_shufflevector(alo, ahi, 0,1,2,3,4,5,6,7,8,9,10,11,12,13,14,15);
    }
    // B 32x16 f16 layout: lane holds column n=l15, K chunks [hi*8..] and [16+hi*8..]
    #pragma unroll
    for (int j = 0; j < 4; ++j) {
      const _Float16* br = &Bs[cur][wn * 64 + j * 16 + l15][0];
      v8h blo = *(const v8h*)(br + hi * 8);
      v8h bhi = *(const v8h*)(br + 16 + hi * 8);
      bf[j] = __builtin_shufflevector(blo, bhi, 0,1,2,3,4,5,6,7,8,9,10,11,12,13,14,15);
    }
    #pragma unroll
    for (int i = 0; i < 2; ++i)
      #pragma unroll
      for (int j = 0; j < 4; ++j)
        acc[i][j] = __builtin_amdgcn_wmma_f32_16x16x32_f16(
            false, af[i], false, bf[j], (short)0, acc[i][j], false, false);
  }

  // C/D layout: VGPR r, lane -> (m = r + 8*hi, n = l15)
  if (FINAL == 0) {
    #pragma unroll
    for (int i = 0; i < 2; ++i) {
      int gmBase = m0 + wm * 32 + i * 16 + hi * 8;
      #pragma unroll
      for (int j = 0; j < 4; ++j) {
        int gn = n0 + wn * 64 + j * 16 + l15;
        float bn = bias[gn], vn = vvec[gn];
        #pragma unroll
        for (int r = 0; r < 8; ++r) {
          float val = fmaxf(acc[i][j][r] + bn, 0.0f) * vn;
          for (int o = 8; o >= 1; o >>= 1) val += __shfl_xor(val, o, 32);
          if (l15 == 0) atomicAdd(&sbuf[gmBase + r], val);
        }
      }
    }
  } else {
    float add2 = mb2[0];
    #pragma unroll
    for (int i = 0; i < 2; ++i) {
      int gmBase = m0 + wm * 32 + i * 16 + hi * 8;
      #pragma unroll
      for (int j = 0; j < 4; ++j) {
        int gn = n0 + wn * 64 + j * 16 + l15;
        #pragma unroll
        for (int r = 0; r < 8; ++r) {
          int gm = gmBase + r;
          float val = acc[i][j][r] + sbuf[gm] + add2;
          if (gn == 0)      val += c01[(size_t)gm * 2 + 0];
          else if (gn == 1) val += c01[(size_t)gm * 2 + 1];
          out[(size_t)gm * 512 + gn] = val;
        }
      }
    }
  }
}

extern "C" void kernel_launch(void* const* d_in, const int* in_sizes, int n_in,
                              void* d_out, int out_size, void* d_ws, size_t ws_size,
                              hipStream_t stream) {
  (void)in_sizes; (void)n_in; (void)out_size; (void)ws_size;
  const float* x   = (const float*)d_in[0];
  const float* t   = (const float*)d_in[1];
  const float* fc  = (const float*)d_in[2];
  const float* qW1 = (const float*)d_in[3];
  const float* qb1 = (const float*)d_in[4];
  const float* qW2 = (const float*)d_in[5];
  const float* qb2 = (const float*)d_in[6];
  const float* cW1 = (const float*)d_in[7];
  const float* cb1 = (const float*)d_in[8];
  const float* cW2 = (const float*)d_in[9];
  const float* cb2 = (const float*)d_in[10];
  const float* tW1 = (const float*)d_in[11];
  const float* tb1 = (const float*)d_in[12];
  const float* tW2 = (const float*)d_in[13];
  const float* tb2 = (const float*)d_in[14];
  const float* hW1 = (const float*)d_in[15];
  const float* hb1 = (const float*)d_in[16];
  const float* hW2 = (const float*)d_in[17];
  const float* hb2 = (const float*)d_in[18];
  float* out = (float*)d_out;

  char* ws = (char*)d_ws;
  size_t off = 0;
  auto wsalloc = [&](size_t bytes) -> void* {
    void* p = ws + off;
    off = (off + bytes + 255) & ~(size_t)255;
    return p;
  };
  _Float16* xh = (_Float16*)wsalloc((size_t)B_ROWS * DDIM * 2);  // 16 MB
  _Float16* Lh = (_Float16*)wsalloc((size_t)DDIM * DDIM * 2);    // 0.5 MB
  _Float16* WT = (_Float16*)wsalloc((size_t)NH * DDIM * 2);      // 1 MB
  float* bb  = (float*)wsalloc((size_t)NH * 4);
  float* vv  = (float*)wsalloc((size_t)NH * 4);
  float* s   = (float*)wsalloc((size_t)B_ROWS * 4);
  float* c01 = (float*)wsalloc((size_t)B_ROWS * 2 * 4);
  float* mb2 = (float*)wsalloc(256);

  (void)hipMemsetAsync(s, 0, (size_t)B_ROWS * 4, stream);

  k_build_L<<<(DDIM * DDIM) / 256, 256, 0, stream>>>(fc, t, Lh);
  k_prep_weights<<<(NH * DDIM) / 256, 256, 0, stream>>>(qW1, cW1, qb1, cb1, qW2, cW2, WT, bb, vv);
  k_mb2<<<1, 256, 0, stream>>>(qb2, cb2, mb2);
  k_cvt_x<<<(B_ROWS * DDIM / 8) / 256, 256, 0, stream>>>(x, xh);
  k_enso<<<B_ROWS / 8, 256, 0, stream>>>(x, tW1, tb1, tW2, tb2, hW1, hb1, hW2, hb2, c01);

  // hidden GEMM (N=1024): accumulates s[row] via fused epilogue
  k_gemm<0><<<dim3(NH / 128, B_ROWS / 128), 256, 0, stream>>>(
      xh, WT, bb, vv, s, nullptr, nullptr, nullptr);
  // final GEMM (N=512): linear term + s + mb2 + ENSO cols -> out
  k_gemm<1><<<dim3(DDIM / 128, B_ROWS / 128), 256, 0, stream>>>(
      xh, Lh, nullptr, nullptr, s, mb2, c01, out);
}
